// FusedGlobalMutualInformationLoss_45921790329224
// MI455X (gfx1250) — matrix operand order repaired and checked by reading, not compiled, not run on hardware
//
#include <hip/hip_runtime.h>

typedef __attribute__((ext_vector_type(16))) _Float16     v16h;
typedef __attribute__((ext_vector_type(8)))  float        v8f;
typedef __attribute__((ext_vector_type(8)))  unsigned int v8u;

#define NBINS   32
#define PRETERM 1922.0f                 // 1/(2*sigma^2), sigma = 0.5/31
#define INV31   0.03225806451612903f    // bin spacing, centers = i/31
#define WGS_PER_BATCH 64
#define WIN     9                       // +/-4 bins: tail weight < exp(-50)
#define LDSTR   34                      // halves per staged row (32 + 2 pad)
#define ROWU    17                      // u32 per staged row
#define NWAVES  8

// ---------------------------------------------------------------- zero scratch
__global__ void mi_zero_kernel(float* __restrict__ w, int n) {
  int i = blockIdx.x * blockDim.x + threadIdx.x;
  if (i < n) w[i] = 0.0f;
}

// ------------------------------------------------- joint histogram via WMMA
// pab[b][i][j] = sum_n wa[n][i] * wb[n][j]   (raw sum; scaled by 1/N later)
// One wave = 32 voxels per K-step. Each lane scatters its voxel's 9-bin
// Parzen window (normalized, f16) into a per-wave LDS slab [image][bin][voxel];
// fragments are then straight ds_load_b32 in the ISA A/B layouts, feeding
// 4x v_wmma_f32_16x16x32_f16 (2x2 tiling of the 32x32 output).
__global__ void __launch_bounds__(256)
mi_joint_hist_wmma(const float* __restrict__ pred,
                   const float* __restrict__ targ,
                   float* __restrict__ pab,
                   int N, int stepsTotal) {
  __shared__ float hist[NBINS * NBINS];
  __shared__ __align__(16) _Float16 stage[NWAVES][2][NBINS * LDSTR];

  const int batch = blockIdx.y;
  const int lane  = threadIdx.x & 31;
  const int wv    = threadIdx.x >> 5;
  const int wave  = (blockIdx.x * blockDim.x + threadIdx.x) >> 5;
  const int wavesPerBatch = (gridDim.x * blockDim.x) >> 5;

  for (int i = threadIdx.x; i < NBINS * NBINS; i += blockDim.x) hist[i] = 0.0f;
  {
    unsigned int* z = (unsigned int*)&stage[wv][0][0];
    for (int i = lane; i < 2 * NBINS * ROWU; i += 32) z[i] = 0u;
  }
  __syncthreads();

  const int g   = lane >> 4;            // half-wave group (ISA fragment layout)
  const int sub = lane & 15;            // A row m / B col n within a 16-tile

  _Float16* waS = &stage[wv][0][0];     // [bin][voxel], row stride LDSTR halves
  _Float16* wbS = &stage[wv][1][0];
  const unsigned int* wa32 = (const unsigned int*)waS;
  const unsigned int* wb32 = (const unsigned int*)wbS;

  v8f acc00 = {}, acc01 = {}, acc10 = {}, acc11 = {};

  const float* pB = pred + (size_t)batch * (size_t)N;
  const float* tB = targ + (size_t)batch * (size_t)N;

  // Grid-stride over K-steps: bounds are wave-uniform -> EXEC all-1s at WMMA
  for (int s = wave; s < stepsTotal; s += wavesPerBatch) {
    const int vbase = s << 5;
    const float p = pB[vbase + lane];
    const float t = tB[vbase + lane];

    int ip = (int)(p * 31.0f + 0.5f); ip = ip < 0 ? 0 : (ip > 31 ? 31 : ip);
    int it = (int)(t * 31.0f + 0.5f); it = it < 0 ? 0 : (it > 31 ? 31 : it);

    // 9-bin Gaussian window + normalization (exact to ~1e-22 vs full 32 bins)
    float wp[WIN], wt[WIN];
    float sp = 0.0f, st = 0.0f;
#pragma unroll
    for (int d = 0; d < WIN; ++d) {
      const int bi = ip - 4 + d;
      const int bj = it - 4 + d;
      const float dp = p - (float)bi * INV31;
      const float dt = t - (float)bj * INV31;
      float ep = __expf(-PRETERM * dp * dp);
      float et = __expf(-PRETERM * dt * dt);
      ep = (bi >= 0 && bi < NBINS) ? ep : 0.0f;
      et = (bj >= 0 && bj < NBINS) ? et : 0.0f;
      wp[d] = ep; sp += ep;
      wt[d] = et; st += et;
    }
    const float rp = 1.0f / sp;
    const float rt = 1.0f / st;

    // Scatter normalized f16 weights into the per-wave slab (column = lane)
#pragma unroll
    for (int d = 0; d < WIN; ++d) {
      const int bi = ip - 4 + d;
      const int bj = it - 4 + d;
      if (bi >= 0 && bi < NBINS) waS[bi * LDSTR + lane] = (_Float16)(wp[d] * rp);
      if (bj >= 0 && bj < NBINS) wbS[bj * LDSTR + lane] = (_Float16)(wt[d] * rt);
    }
    // DS ops from one wave are in-order; this blocks compiler reordering and
    // covers the staged-store -> fragment-load dependency.
    asm volatile("s_wait_dscnt 0x0" ::: "memory");

    // Fragment gather per the CDNA5 ISA VGPR layouts (CDNA5 ISA):
    //  A 16x32 f16: elem e -> K = (e&8?16:0) + 8g + (e&7), row m = sub (+0/16)
    //  B 32x16 f16: elem e -> K = (e&8?16:0) + 8g + (e&7), col n = sub (+0/16)
    // Consecutive-voxel half pairs at fixed bin == one b32 load.
    v8u ua0, ua1, ub0, ub1;
#pragma unroll
    for (int v = 0; v < 8; ++v) {
      const int kA2 = ((v >= 4) ? 8 : 0) + 4 * g + (v & 3);  // K_A/2
      const int kB2 = ((v >= 4) ? 8 : 0) + 4 * g + (v & 3);  // K_B/2
      ua0[v] = wa32[(sub)      * ROWU + kA2];
      ua1[v] = wa32[(sub + 16) * ROWU + kA2];
      ub0[v] = wb32[(sub)      * ROWU + kB2];
      ub1[v] = wb32[(sub + 16) * ROWU + kB2];
    }
    asm volatile("" ::: "memory");

    // Re-zero only this iteration's window (slab stays zero elsewhere)
#pragma unroll
    for (int d = 0; d < WIN; ++d) {
      const int bi = ip - 4 + d;
      const int bj = it - 4 + d;
      if (bi >= 0 && bi < NBINS) waS[bi * LDSTR + lane] = (_Float16)0.0f;
      if (bj >= 0 && bj < NBINS) wbS[bj * LDSTR + lane] = (_Float16)0.0f;
    }

    const v16h a0 = __builtin_bit_cast(v16h, ua0);
    const v16h a1 = __builtin_bit_cast(v16h, ua1);
    const v16h b0 = __builtin_bit_cast(v16h, ub0);
    const v16h b1 = __builtin_bit_cast(v16h, ub1);

    // D = A x B + C : pab[m][n] += sum_k wa[k][m]*wb[k][n]
    acc00 = __builtin_amdgcn_wmma_f32_16x16x32_f16(false, a0, false, b0, (short)0, acc00, false, false);
    acc01 = __builtin_amdgcn_wmma_f32_16x16x32_f16(false, a0, false, b1, (short)0, acc01, false, false);
    acc10 = __builtin_amdgcn_wmma_f32_16x16x32_f16(false, a1, false, b0, (short)0, acc10, false, false);
    acc11 = __builtin_amdgcn_wmma_f32_16x16x32_f16(false, a1, false, b1, (short)0, acc11, false, false);
  }

  // C/D layout: lane L, VGPR r -> element (M = r + 8*(L/16), N = L%16)
#pragma unroll
  for (int r = 0; r < 8; ++r) {
    const int M = r + (g << 3);
    atomicAdd(&hist[M * NBINS + sub],             acc00[r]);
    atomicAdd(&hist[M * NBINS + sub + 16],        acc01[r]);
    atomicAdd(&hist[(M + 16) * NBINS + sub],      acc10[r]);
    atomicAdd(&hist[(M + 16) * NBINS + sub + 16], acc11[r]);
  }
  __syncthreads();

  float* dst = pab + (size_t)batch * NBINS * NBINS;
  for (int i = threadIdx.x; i < NBINS * NBINS; i += blockDim.x)
    atomicAdd(&dst[i], hist[i]);
}

// ------------------------------------------------------------------ finalize
// Marginals are exact row/col sums of pab (weights sum to 1 per voxel).
__global__ void __launch_bounds__(32)
mi_finalize(const float* __restrict__ pab, float* __restrict__ out,
            int B, float invN) {
  const int lane = threadIdx.x;   // lane = histogram row i
  float mi_sum = 0.0f;
  for (int b = 0; b < B; ++b) {
    const float* P = pab + (size_t)b * NBINS * NBINS;
    float row[NBINS];
    float pa = 0.0f;
#pragma unroll
    for (int j = 0; j < NBINS; ++j) {
      row[j] = P[lane * NBINS + j] * invN;
      pa += row[j];
    }
    float pb = 0.0f;
#pragma unroll
    for (int i = 0; i < NBINS; ++i) pb += P[i * NBINS + lane];
    pb *= invN;

    float part = 0.0f;
#pragma unroll
    for (int j = 0; j < NBINS; ++j) {
      const float pbj  = __shfl(pb, j, 32);
      const float pij  = row[j];
      const float papb = pa * pbj;
      part += pij * __logf((pij + 1e-7f) / (papb + 1e-7f) + 1e-7f);
    }
    for (int off = 16; off > 0; off >>= 1) part += __shfl_down(part, off, 32);
    if (lane == 0) mi_sum += part;
  }
  if (lane == 0) out[0] = -mi_sum / (float)B;
}

extern "C" void kernel_launch(void* const* d_in, const int* in_sizes, int n_in,
                              void* d_out, int out_size, void* d_ws, size_t ws_size,
                              hipStream_t stream) {
  const float* pred = (const float*)d_in[0];
  const float* targ = (const float*)d_in[1];
  float* out = (float*)d_out;
  float* pab = (float*)d_ws;            // 2 * 32 * 32 floats = 8 KB scratch

  const int B = 2;
  const int total = in_sizes[0];        // 2 * 96^3
  const int N = total / B;              // voxels per batch (divisible by 32)
  const int stepsTotal = N / 32;

  const int histElems = B * NBINS * NBINS;
  mi_zero_kernel<<<(histElems + 255) / 256, 256, 0, stream>>>(pab, histElems);

  dim3 grid(WGS_PER_BATCH, B);
  mi_joint_hist_wmma<<<grid, 256, 0, stream>>>(pred, targ, pab, N, stepsTotal);

  mi_finalize<<<1, 32, 0, stream>>>(pab, out, B, 1.0f / (float)N);
}
